// ST_GCN_ALN74_42958262895288
// MI455X (gfx1250) — matrix operand order repaired and checked by reading, not compiled, not run on hardware
//
#include <hip/hip_runtime.h>
#include <hip/hip_bf16.h>

typedef __attribute__((ext_vector_type(16))) _Float16 v16h;
typedef __attribute__((ext_vector_type(8)))  float    v8f;

#define EPS 1e-5f

// ---------------------------------------------------------------------------
// WMMA helper (CDNA5: V_WMMA_F32_16X16X32_F16, wave32)
// ---------------------------------------------------------------------------
__device__ __forceinline__ v8f wmma16(const v16h& a, const v16h& b, const v8f& c) {
    return __builtin_amdgcn_wmma_f32_16x16x32_f16(false, a, false, b, (short)0, c, false, false);
}

// Fragment element placement (ISA §7.12.2, 16-bit layouts):
// A (16x32): (row,K) -> lane = row + ((K>>3)&1)*16 ; slot = (K&7) + (K>=16 ? 8 : 0)
// B (32x16): (K,col) -> lane = col + ((K>>3)&1)*16 ; slot = (K&7) + (K>=16 ? 8 : 0)
// C/D f32  : acc[r]  -> row = r + (lane>=16 ? 8 : 0), col = lane&15
__device__ __forceinline__ int lds_off_a(int mtile, int r, int kk) {
    int lane = (r & 15) + (((kk >> 3) & 1) << 4);
    int slot = (kk & 7) + ((kk & 16) ? 8 : 0);
    return (mtile * 32 + lane) * 16 + slot;
}
__device__ __forceinline__ int lds_off_b(int ntile, int kk, int col) {
    int lane = (col & 15) + (((kk >> 3) & 1) << 4);
    int slot = (kk & 7) + ((kk & 16) ? 8 : 0);
    return (ntile * 32 + lane) * 16 + slot;
}

// ---------------------------------------------------------------------------
// Generic WMMA GEMM:  D[b] = act( A[b](MxK) * B[b](KxN) + bias , BN )
// Block 256 thr = 8 waves (2 wave-rows x 4 wave-cols); wave tile 32x32 (2x2
// fragments, 4 accumulators) -> block tile 64x128, K-step 32.
// Staging: fixed-trip unrolled, 32-bit offsets, invariants hoisted.
// ---------------------------------------------------------------------------
__global__ __launch_bounds__(256)
void gemm_wmma_kernel(const _Float16* __restrict__ A, long long sA, int lda, int M,
                      const _Float16* __restrict__ B, long long sB, int ldb, int Kd, int N,
                      int bTrans, int colGrp, int colGrpStride,
                      const float* __restrict__ biasRow, const float* __restrict__ biasCol,
                      const float* __restrict__ bnG, const float* __restrict__ bnB,
                      const float* __restrict__ bnM, const float* __restrict__ bnV,
                      int relu,
                      float* __restrict__ Df, _Float16* __restrict__ Dh,
                      long long sD, int ldd)
{
    const int bz = blockIdx.z;
    const _Float16* Ab = A + (long long)bz * sA;
    const _Float16* Bb = B + (long long)bz * sB;
    const int row0 = blockIdx.y * 64;
    const int col0 = blockIdx.x * 128;

    __shared__ v16h Afrag[4 * 32];    // [mtile][lane]
    __shared__ v16h Bfrag[8 * 32];    // [ntile][lane]
    _Float16* Ah = (_Float16*)Afrag;
    _Float16* Bh = (_Float16*)Bfrag;

    const int tid  = threadIdx.x;
    const int wave = tid >> 5, lane = tid & 31;
    const int wm = wave & 1, wn = wave >> 1;

    // ---- per-thread staging invariants (independent of k0) ----
    int aBase[8], aLds[8], aKK[8];                 // A: 64x32 -> 8 elems/thread
    int aRowOk = 0;
#pragma unroll
    for (int it = 0; it < 8; ++it) {
        int i = tid + it * 256;
        int r = i >> 5, kk = i & 31;
        int gr = row0 + r;
        if (gr < M) aRowOk |= (1 << it);
        int grc = gr < M ? gr : (M - 1);
        aBase[it] = grc * lda + kk;
        aKK[it]   = kk;
        aLds[it]  = lds_off_a(r >> 4, r, kk);
    }
    const int bStep = bTrans ? 1 : ldb;
    int bBase[16], bLds[16], bKK[16];              // B: 32x128 -> 16 elems/thread
    int bColOk = 0;
#pragma unroll
    for (int it = 0; it < 16; ++it) {
        int i = tid + it * 256;
        int kk = i >> 7, j = i & 127;
        int gc = col0 + j;
        if (gc < N) bColOk |= (1 << it);
        int gcc = gc < N ? gc : (N - 1);
        int base;
        if (bTrans) {
            base = gcc * ldb + kk;
        } else {
            int cpart = colGrp ? ((gcc / colGrp) * colGrpStride + (gcc % colGrp)) : gcc;
            base = kk * ldb + cpart;
        }
        bBase[it] = base;
        bKK[it]   = kk;
        bLds[it]  = lds_off_b(j >> 4, kk, j);
    }

    v8f acc[4] = {{}, {}, {}, {}};                 // [mi*2+ni]

    for (int k0 = 0; k0 < Kd; k0 += 32) {
        if (k0 + 32 < Kd) {
            int pr = row0 + (tid & 63);
            pr = pr < M - 1 ? pr : M - 1;
            __builtin_prefetch(Ab + pr * lda + (k0 + 32), 0, 1);
        }
#pragma unroll
        for (int it = 0; it < 8; ++it) {
            bool ok = ((aRowOk >> it) & 1) && (k0 + aKK[it] < Kd);
            _Float16 v = Ab[ok ? (aBase[it] + k0) : 0];
            Ah[aLds[it]] = ok ? v : (_Float16)0.f;
        }
        const int kOff = k0 * bStep;
#pragma unroll
        for (int it = 0; it < 16; ++it) {
            bool ok = ((bColOk >> it) & 1) && (k0 + bKK[it] < Kd);
            _Float16 v = Bb[ok ? (bBase[it] + kOff) : 0];
            Bh[bLds[it]] = ok ? v : (_Float16)0.f;
        }
        __syncthreads();
        v16h a0 = Afrag[(wm * 2 + 0) * 32 + lane];
        v16h a1 = Afrag[(wm * 2 + 1) * 32 + lane];
        v16h b0 = Bfrag[(wn * 2 + 0) * 32 + lane];
        v16h b1 = Bfrag[(wn * 2 + 1) * 32 + lane];
        acc[0] = wmma16(a0, b0, acc[0]);
        acc[1] = wmma16(a0, b1, acc[1]);
        acc[2] = wmma16(a1, b0, acc[2]);
        acc[3] = wmma16(a1, b1, acc[3]);
        __syncthreads();
    }

    const int rowL = ((lane >> 4) & 1) * 8;
    const int colL = lane & 15;
#pragma unroll
    for (int mi = 0; mi < 2; ++mi) {
#pragma unroll
        for (int ni = 0; ni < 2; ++ni) {
            const v8f& ac = acc[mi * 2 + ni];
            int cb = col0 + (wn * 2 + ni) * 16 + colL;
            if (cb >= N) continue;
            int rb = row0 + (wm * 2 + mi) * 16 + rowL;
#pragma unroll
            for (int r = 0; r < 8; ++r) {
                int grow = rb + r;
                if (grow >= M) continue;
                float v = ac[r];
                if (biasRow) v += biasRow[grow];
                if (biasCol) v += biasCol[cb];
                if (bnG) {
                    float sc = bnG[grow] * __frsqrt_rn(bnV[grow] + EPS);
                    v = (v - bnM[grow]) * sc + bnB[grow];
                }
                if (relu) v = v > 0.f ? v : 0.f;
                long long o = (long long)bz * sD + grow * ldd + cb;
                if (Df) Df[o] = v;
                if (Dh) Dh[o] = (_Float16)v;
            }
        }
    }
}

// ---------------------------------------------------------------------------
// Adjacency contraction + fused BN1 + ReLU:
//   z[n,c,t,w] = relu(BN1( sum_{k,v} y[n,k,c,t,v] * (0.5*A[k]*imp[k] + 0.5*B[n,k])[v,w] ))
// rows = cout*Tin is always a multiple of 16 -> no row guards on A loads.
// Blend matrices built once per workgroup in LDS, directly in B-fragment layout.
// ---------------------------------------------------------------------------
__global__ __launch_bounds__(256)
void adj_bn_relu_kernel(const _Float16* __restrict__ Y,      // [n][3][rows*25]
                        const float* __restrict__ Abase,     // [3][25][25]
                        const float* __restrict__ Imp,       // [3][25][25]
                        const float* __restrict__ Bmat,      // [n][3][25][25]
                        const float* __restrict__ g, const float* __restrict__ b,
                        const float* __restrict__ m, const float* __restrict__ vv,
                        _Float16* __restrict__ Z,            // [n][rows*25]
                        int cout, int Tin)
{
    const int n    = blockIdx.y;
    const int rows = cout * Tin;
    __shared__ v16h Mfrag[3 * 2 * 32];   // [k][ntile][lane]
    _Float16* Mh = (_Float16*)Mfrag;

    const int tid = threadIdx.x;
    const float* BmN = Bmat + n * 1875;
#pragma unroll
    for (int it = 0; it < 6; ++it) {                  // 6 trips x 256 = 1536 of the 3*2*32*16 halves
        int i = tid + it * 256;
        int slot = i & 15;
        int vec  = i >> 4;
        int lane2 = vec & 31;
        int tile  = (vec >> 5) & 1;
        int k     = vec / 64;
        int K   = ((lane2 & 16) ? 8 : 0) + (slot & 7) + ((slot & 8) ? 16 : 0);     // v (reduction) index
        int col = tile * 16 + (lane2 & 15);           // w index
        float val = 0.f;
        if (K < 25 && col < 25) {
            int idx = k * 625 + K * 25 + col;
            val = 0.5f * Abase[idx] * Imp[idx] + 0.5f * BmN[idx];
        }
        Mh[i] = (_Float16)val;
    }
    __syncthreads();

    const int wave = tid >> 5, lane = tid & 31;
    const int rowBase = (blockIdx.x * 8 + wave) * 16;
    if (rowBase >= rows) return;  // wave-uniform

    const _Float16* Yn = Y + (long long)n * 3 * rows * 25;
    const int arow = rowBase + (lane & 15);
    const int kb   = (lane < 16) ? 0 : 8;

    v8f acc0 = {}, acc1 = {};
#pragma unroll
    for (int k = 0; k < 3; ++k) {
        const _Float16* Yr = Yn + (k * rows + arow) * 25;
        v16h af;
#pragma unroll
        for (int i = 0; i < 8; ++i) {
            af[i] = Yr[kb + i];                     // K = kb+i <= 15 < 25: always valid
            int kk1 = 16 + kb + i;                  // K in [16..31]: valid iff < 25
            _Float16 t = Yr[kk1 <= 24 ? kk1 : 24];  // clamped load + select (no branch)
            af[8 + i] = (kk1 < 25) ? t : (_Float16)0.f;
        }
        v16h bf0 = Mfrag[(k * 2 + 0) * 32 + lane];
        v16h bf1 = Mfrag[(k * 2 + 1) * 32 + lane];
        acc0 = wmma16(af, bf0, acc0);
        acc1 = wmma16(af, bf1, acc1);
    }

    _Float16* Zn = Z + (long long)n * rows * 25;
    const int rowL = ((lane >> 4) & 1) * 8;
    const int colL = lane & 15;
#pragma unroll
    for (int r = 0; r < 8; ++r) {
        int grow = rowBase + rowL + r;
        int d = grow / Tin;
        float sc = g[d] * __frsqrt_rn(vv[d] + EPS);
        float sh = b[d] - m[d] * sc;
#pragma unroll
        for (int j = 0; j < 2; ++j) {
            int w = colL + j * 16;
            if (w < 25) {
                float z = (j ? acc1[r] : acc0[r]) * sc + sh;
                z = z > 0.f ? z : 0.f;
                Zn[grow * 25 + w] = (_Float16)z;
            }
        }
    }
}

// ---------------------------------------------------------------------------
// TCN implicit GEMM (9x1 conv, pad 4, temporal stride s) + bias + BN2 +
// residual + ReLU.  Reduction r=(c,dt): c=r/9, dt=r%9, t = s*t'+dt-4.
// Block tile 64x128, 8 waves (2x4), wave tile 32x32 (2x2 frags, 4 acc).
// ---------------------------------------------------------------------------
__global__ __launch_bounds__(256)
void tcn_kernel(const _Float16* __restrict__ Wt,   // [cout][cin*9]
                const _Float16* __restrict__ Zin,  // [n][cin*Tin*25]
                const float* __restrict__ bias,
                const float* __restrict__ g, const float* __restrict__ b,
                const float* __restrict__ m, const float* __restrict__ vv,
                const _Float16* __restrict__ Res,  // null | [n][cout*Tout*25]
                _Float16* __restrict__ Out,        // [n][cout*Tout*25]
                int cout, int cin, int Tin, int Tout, int stride)
{
    const int n  = blockIdx.z;
    const int R  = cin * 9;
    const int N  = Tout * 25;
    const int Tin25 = Tin * 25;
    const int row0 = blockIdx.y * 64;
    const int col0 = blockIdx.x * 128;

    __shared__ v16h Afrag[4 * 32];   // [mtile][lane]
    __shared__ v16h Bfrag[8 * 32];   // [ntile][lane]
    _Float16* Ah = (_Float16*)Afrag;
    _Float16* Bh = (_Float16*)Bfrag;

    const int tid = threadIdx.x, wave = tid >> 5, lane = tid & 31;
    const int wm = wave & 1, wn = wave >> 1;
    const _Float16* Zn = Zin + (long long)n * cin * Tin25;

    // ---- per-thread staging invariants ----
    int aBase[8], aLds[8], aKK[8];
    int aRowOk = 0;
#pragma unroll
    for (int it = 0; it < 8; ++it) {                  // 64*32/256
        int i = tid + it * 256;
        int r = i >> 5, kk = i & 31;
        int gr = row0 + r;
        if (gr < cout) aRowOk |= (1 << it);
        int grc = gr < cout ? gr : (cout - 1);
        aBase[it] = grc * R + kk;
        aKK[it]   = kk;
        aLds[it]  = lds_off_a(r >> 4, r, kk);
    }
    int bColAdd[16], bLds[16], bKK[16], bTB[16];      // colAdd = (s*t'-4)*25 + v
    int bColOk = 0;
#pragma unroll
    for (int it = 0; it < 16; ++it) {                 // 32*128/256
        int i = tid + it * 256;
        int kk = i >> 7, j = i & 127;
        int gc = col0 + j;
        if (gc < N) bColOk |= (1 << it);
        int gcc = gc < N ? gc : (N - 1);
        int tp = gcc / 25, vj = gcc - 25 * tp;
        int tb = stride * tp - 4;                     // t = tb + dt
        bTB[it]     = tb;
        bColAdd[it] = tb * 25 + vj;
        bKK[it]     = kk;
        bLds[it]    = lds_off_b(j >> 4, kk, j);
    }

    v8f acc[4] = {{}, {}, {}, {}};
    for (int k0 = 0; k0 < R; k0 += 32) {
        if (k0 + 32 < R) {
            int pr = row0 + (tid & 63);
            pr = pr < cout - 1 ? pr : cout - 1;
            __builtin_prefetch(Wt + pr * R + (k0 + 32), 0, 1);
        }
#pragma unroll
        for (int it = 0; it < 8; ++it) {
            bool ok = ((aRowOk >> it) & 1) && (k0 + aKK[it] < R);
            _Float16 v = Wt[ok ? (aBase[it] + k0) : 0];
            Ah[aLds[it]] = ok ? v : (_Float16)0.f;
        }
#pragma unroll
        for (int it = 0; it < 16; ++it) {
            int gk = k0 + bKK[it];
            int c  = gk / 9, dt = gk - 9 * c;
            int t  = bTB[it] + dt;
            bool ok = ((bColOk >> it) & 1) && (gk < R) && (t >= 0) && (t < Tin);
            int off = c * Tin25 + dt * 25 + bColAdd[it];
            _Float16 v = Zn[ok ? off : 0];
            Bh[bLds[it]] = ok ? v : (_Float16)0.f;
        }
        __syncthreads();
        v16h a0 = Afrag[(wm * 2 + 0) * 32 + lane];
        v16h a1 = Afrag[(wm * 2 + 1) * 32 + lane];
        v16h b0 = Bfrag[(wn * 2 + 0) * 32 + lane];
        v16h b1 = Bfrag[(wn * 2 + 1) * 32 + lane];
        acc[0] = wmma16(a0, b0, acc[0]);
        acc[1] = wmma16(a0, b1, acc[1]);
        acc[2] = wmma16(a1, b0, acc[2]);
        acc[3] = wmma16(a1, b1, acc[3]);
        __syncthreads();
    }

    const int rowL = ((lane >> 4) & 1) * 8;
    const int colL = lane & 15;
    const _Float16* ResN = Res ? (Res + (long long)n * cout * N) : nullptr;
    _Float16* OutN = Out + (long long)n * cout * N;
#pragma unroll
    for (int mi = 0; mi < 2; ++mi) {
#pragma unroll
        for (int ni = 0; ni < 2; ++ni) {
            const v8f& ac = acc[mi * 2 + ni];
            int cb = col0 + (wn * 2 + ni) * 16 + colL;
            if (cb >= N) continue;
            int rb = row0 + (wm * 2 + mi) * 16 + rowL;
#pragma unroll
            for (int r = 0; r < 8; ++r) {
                int d = rb + r;
                if (d >= cout) continue;
                float sc = g[d] * __frsqrt_rn(vv[d] + EPS);
                float v0 = ac[r] + bias[d];
                v0 = (v0 - m[d]) * sc + b[d];
                int o = d * N + cb;
                if (ResN) v0 += (float)ResN[o];
                v0 = v0 > 0.f ? v0 : 0.f;
                OutN[o] = (_Float16)v0;
            }
        }
    }
}

// ---------------------------------------------------------------------------
// Small elementwise / reduction kernels
// ---------------------------------------------------------------------------
__global__ void f32_to_f16_kernel(const float* __restrict__ in, _Float16* __restrict__ out, long long count) {
    long long i = (long long)blockIdx.x * blockDim.x + threadIdx.x;
    if (i < count) out[i] = (_Float16)in[i];
}

__global__ void data_bn_kernel(const float* __restrict__ x,
                               const float* __restrict__ g, const float* __restrict__ b,
                               const float* __restrict__ m, const float* __restrict__ vv,
                               _Float16* __restrict__ h, _Float16* __restrict__ hT, int total)
{
    int i = blockIdx.x * blockDim.x + threadIdx.x;
    if (i >= total) return;
    int vI = i % 25;
    int t  = (i / 25) % 300;
    int c  = (i / (25 * 300)) % 3;
    int nn = i / (25 * 300 * 3);
    int ns = nn >> 1, mm = nn & 1;
    int xi = ((((ns * 3 + c) * 300 + t) * 25 + vI) << 1) + mm;
    int ch = vI * 3 + c;
    float sc  = g[ch] * __frsqrt_rn(vv[ch] + EPS);
    float val = (x[xi] - m[ch]) * sc + b[ch];
    h[i] = (_Float16)val;
    hT[(nn * 300 + t) * 75 + c * 25 + vI] = (_Float16)val;
}

__global__ void bnrelu_kernel(const float* __restrict__ in,
                              const float* __restrict__ g, const float* __restrict__ b,
                              const float* __restrict__ m, const float* __restrict__ vv,
                              _Float16* __restrict__ out, int total, int nch)
{
    int i = blockIdx.x * blockDim.x + threadIdx.x;
    if (i >= total) return;
    int ch = i % nch;
    float sc = g[ch] * __frsqrt_rn(vv[ch] + EPS);
    float v  = (in[i] - m[ch]) * sc + b[ch];
    out[i] = (_Float16)(v > 0.f ? v : 0.f);
}

__global__ void softmax_gate_kernel(const float* __restrict__ y3, float* __restrict__ Bm, int nrows)
{
    int i = blockIdx.x * blockDim.x + threadIdx.x;
    if (i >= nrows) return;
    const float* r = y3 + i * 25;
    float mx = -1e30f;
#pragma unroll
    for (int j = 0; j < 25; ++j) mx = fmaxf(mx, r[j]);
    float e[25], s = 0.f;
#pragma unroll
    for (int j = 0; j < 25; ++j) { e[j] = __expf(r[j] - mx); s += e[j]; }
    float inv = 1.f / s;
#pragma unroll
    for (int j = 0; j < 25; ++j) {
        float y = e[j] * inv;
        float mult = (y > 0.1f) ? 1.f : ((y < 0.1f) ? 0.001f : y);
        Bm[i * 25 + j] = mult * y;
    }
}

__global__ void pool_kernel(const _Float16* __restrict__ act, float* __restrict__ pooled, int C, int TV)
{
    int i = blockIdx.x * blockDim.x + threadIdx.x;   // i over 32*C
    if (i >= 32 * C) return;
    int nn = i / C, c = i % C;
    const _Float16* p = act + (nn * C + c) * TV;
    float s = 0.f;
    for (int j = 0; j < TV; ++j) s += (float)p[j];
    pooled[i] = s / (float)TV;
}

__global__ void fc_kernel(const float* __restrict__ pooled, const float* __restrict__ fw,
                          const float* __restrict__ fb, float* __restrict__ out)
{
    int i = blockIdx.x * blockDim.x + threadIdx.x;   // 16*60
    if (i >= 16 * 60) return;
    int ns = i / 60, j = i % 60;
    const float* p0 = pooled + (2 * ns) * 256;
    const float* p1 = pooled + (2 * ns + 1) * 256;
    const float* w  = fw + j * 256;
    float s = 0.f;
    for (int c = 0; c < 256; ++c) s += 0.5f * (p0[c] + p1[c]) * w[c];
    out[ns * 60 + j] = s + fb[j];
}

// ---------------------------------------------------------------------------
// Host orchestration
// ---------------------------------------------------------------------------
struct BlockCfg { int cin, cout, stride, res; };           // res: 0 none, 1 iden, 2 conv
static const BlockCfg CFGH[10] = {
    {3, 64, 1, 0}, {64, 64, 1, 1}, {64, 64, 1, 1}, {64, 64, 1, 1},
    {64, 128, 2, 2}, {128, 128, 1, 1}, {128, 128, 1, 1},
    {128, 256, 2, 2}, {256, 256, 1, 1}, {256, 256, 1, 1}};

static inline int ceil_div(int a, int b) { return (a + b - 1) / b; }

extern "C" void kernel_launch(void* const* d_in, const int* in_sizes, int n_in,
                              void* d_out, int out_size, void* d_ws, size_t ws_size,
                              hipStream_t stream)
{
    (void)in_sizes; (void)n_in; (void)out_size; (void)ws_size;
    int p = 0;
    auto F = [&](void) -> const float* { return (const float*)d_in[p++]; };
    const float* X = F(); const float* Abase = F();
    const float *dg = F(), *db = F(), *dm = F(), *dv = F();
    const float *c1w = F(), *c1b = F();
    const float *ag = F(), *ab = F(), *am = F(), *av = F();
    const float *w1 = F(), *b1 = F(), *w2 = F(), *b2 = F(), *w3 = F(), *b3 = F();
    struct BP { const float *wg,*bg,*g1,*be1,*m1,*v1,*wt,*bt,*g2,*be2,*m2,*v2,*rw,*rb,*rg,*rbe,*rm,*rv; } bp[10];
    for (int i = 0; i < 10; ++i) {
        bp[i].wg = F(); bp[i].bg = F();
        bp[i].g1 = F(); bp[i].be1 = F(); bp[i].m1 = F(); bp[i].v1 = F();
        bp[i].wt = F(); bp[i].bt = F();
        bp[i].g2 = F(); bp[i].be2 = F(); bp[i].m2 = F(); bp[i].v2 = F();
        if (CFGH[i].res == 2) {
            bp[i].rw = F(); bp[i].rb = F();
            bp[i].rg = F(); bp[i].rbe = F(); bp[i].rm = F(); bp[i].rv = F();
        } else { bp[i].rw = bp[i].rb = bp[i].rg = bp[i].rbe = bp[i].rm = bp[i].rv = nullptr; }
    }
    const float* edge = F(); const float* fw = F(); const float* fb = F();

    // deterministic bump allocator over d_ws
    char* wsp = (char*)d_ws;
    auto alloc = [&](size_t bytes) -> void* {
        void* r = (void*)wsp; wsp += (bytes + 255) & ~(size_t)255; return r; };

    const long long ACT = 32LL * 64 * 300 * 25;            // 15.36M elems (max per stage)
    _Float16* actA = (_Float16*)alloc(ACT * 2);
    _Float16* actB = (_Float16*)alloc(ACT * 2);
    _Float16* zh   = (_Float16*)alloc(ACT * 2);
    _Float16* rres = (_Float16*)alloc(ACT * 2);
    _Float16* ygcn = (_Float16*)alloc(3 * ACT * 2);
    _Float16* hT   = (_Float16*)alloc(32LL * 300 * 75 * 2);
    float*    y0   = (float*)alloc(32LL * 375 * 4);
    _Float16* a0   = (_Float16*)alloc(32LL * 375 * 2);
    _Float16* a1   = (_Float16*)alloc(32LL * 1500 * 2);
    _Float16* a2   = (_Float16*)alloc(32LL * 1500 * 2);
    float*    y3   = (float*)alloc(32LL * 1875 * 4);
    float*    Bm   = (float*)alloc(32LL * 1875 * 4);
    float*    pooled = (float*)alloc(32LL * 256 * 4);

    auto conv16 = [&](const float* src, long long n) -> _Float16* {
        _Float16* d = (_Float16*)alloc(n * 2);
        f32_to_f16_kernel<<<dim3((unsigned)((n + 255) / 256)), 256, 0, stream>>>(src, d, n);
        return d;
    };

    _Float16* c1wh = conv16(c1w, 5LL * 300);
    _Float16* w1h  = conv16(w1, 1500LL * 375);
    _Float16* w2h  = conv16(w2, 1500LL * 1500);
    _Float16* w3h  = conv16(w3, 1875LL * 1500);
    _Float16 *wgh[10], *wth[10], *rwh[10];
    for (int i = 0; i < 10; ++i) {
        const BlockCfg& c = CFGH[i];
        wgh[i] = conv16(bp[i].wg, 3LL * c.cout * c.cin);
        wth[i] = conv16(bp[i].wt, (long long)c.cout * c.cin * 9);
        rwh[i] = (c.res == 2) ? conv16(bp[i].rw, (long long)c.cout * c.cin) : nullptr;
    }

    // stage 0: data BN -> actA (f16) + hT (f16)
    {
        int total = 32 * 3 * 300 * 25;
        data_bn_kernel<<<dim3(ceil_div(total, 256)), 256, 0, stream>>>(X, dg, db, dm, dv, actA, hT, total);
    }

    // ANet
    gemm_wmma_kernel<<<dim3(1, 1, 32), 256, 0, stream>>>(
        c1wh, 0, 300, 5, hT, 300LL * 75, 75, 300, 75,
        0, 0, 0, c1b, nullptr, nullptr, nullptr, nullptr, nullptr, 0,
        y0, nullptr, 375, 75);
    bnrelu_kernel<<<dim3(ceil_div(32 * 375, 256)), 256, 0, stream>>>(y0, ag, ab, am, av, a0, 32 * 375, 375);
    gemm_wmma_kernel<<<dim3(ceil_div(1500, 128), 1, 1), 256, 0, stream>>>(
        a0, 0, 375, 32, w1h, 0, 375, 375, 1500,
        1, 0, 0, nullptr, b1, nullptr, nullptr, nullptr, nullptr, 1,
        nullptr, a1, 0, 1500);
    gemm_wmma_kernel<<<dim3(ceil_div(1500, 128), 1, 1), 256, 0, stream>>>(
        a1, 0, 1500, 32, w2h, 0, 1500, 1500, 1500,
        1, 0, 0, nullptr, b2, nullptr, nullptr, nullptr, nullptr, 1,
        nullptr, a2, 0, 1500);
    gemm_wmma_kernel<<<dim3(ceil_div(1875, 128), 1, 1), 256, 0, stream>>>(
        a2, 0, 1500, 32, w3h, 0, 1500, 1500, 1875,
        1, 0, 0, nullptr, b3, nullptr, nullptr, nullptr, nullptr, 0,
        y3, nullptr, 0, 1875);
    softmax_gate_kernel<<<dim3(ceil_div(32 * 75, 128)), 128, 0, stream>>>(y3, Bm, 32 * 75);

    // 10 st-gcn blocks
    _Float16* cur = actA;
    _Float16* nxt = actB;
    int Tin = 300;
    for (int i = 0; i < 10; ++i) {
        const BlockCfg& c = CFGH[i];
        const int Tout = (Tin - 1) / c.stride + 1;
        const int Nin  = Tin * 25;
        const int Nout = Tout * 25;
        const int Mg   = 3 * c.cout;

        gemm_wmma_kernel<<<dim3(ceil_div(Nin, 128), ceil_div(Mg, 64), 32), 256, 0, stream>>>(
            wgh[i], 0, c.cin, Mg, cur, (long long)c.cin * Nin, Nin, c.cin, Nin,
            0, 0, 0, bp[i].bg, nullptr, nullptr, nullptr, nullptr, nullptr, 0,
            nullptr, ygcn, (long long)Mg * Nin, Nin);

        {
            int rows = c.cout * Tin;
            adj_bn_relu_kernel<<<dim3(ceil_div(rows, 128), 32), 256, 0, stream>>>(
                ygcn, Abase, edge + (long long)i * 1875, Bm,
                bp[i].g1, bp[i].be1, bp[i].m1, bp[i].v1, zh, c.cout, Tin);
        }

        const _Float16* resPtr = nullptr;
        if (c.res == 1) resPtr = cur;
        else if (c.res == 2) {
            gemm_wmma_kernel<<<dim3(ceil_div(Nout, 128), ceil_div(c.cout, 64), 32), 256, 0, stream>>>(
                rwh[i], 0, c.cin, c.cout, cur, (long long)c.cin * Nin, Nin, c.cin, Nout,
                0, 25, c.stride * 25, bp[i].rb, nullptr,
                bp[i].rg, bp[i].rbe, bp[i].rm, bp[i].rv, 0,
                nullptr, rres, (long long)c.cout * Nout, Nout);
            resPtr = rres;
        }

        tcn_kernel<<<dim3(ceil_div(Nout, 128), ceil_div(c.cout, 64), 32), 256, 0, stream>>>(
            wth[i], zh, bp[i].bt, bp[i].g2, bp[i].be2, bp[i].m2, bp[i].v2,
            resPtr, nxt, c.cout, c.cout, Tin, Tout, c.stride);

        _Float16* tmp = cur; cur = nxt; nxt = tmp;
        Tin = Tout;
    }

    pool_kernel<<<dim3(ceil_div(32 * 256, 256)), 256, 0, stream>>>(cur, pooled, 256, Tin * 25);
    fc_kernel<<<dim3(ceil_div(16 * 60, 128)), 128, 0, stream>>>(pooled, fw, fb, (float*)d_out);
}
